// DualMemoryCell_53618371723294
// MI455X (gfx1250) — matrix operand order repaired and checked by reading, not compiled, not run on hardware
//
#include <hip/hip_runtime.h>

// ---------------------------------------------------------------------------
// DualMemoryCell for MI455X (gfx1250, wave32, WMMA 16x16x32 f16 -> f32 acc)
//   node 1: init (weight f16 convert, state copy, barrier reset)
//   node 2: hoisted xw = x_seq @ W_x^T + b_h   (65536x512x512 WMMA GEMM)
//   node 3: persistent scan kernel, 32 WGs, grid-barrier phased, T=2048 steps
// ---------------------------------------------------------------------------

#define DIMD   512
#define TLEN   2048
#define BATCH  32
#define NSLOT  32
#define NWG    32
#define INV_SQRT_D 0.04419417382415922f   // 1/sqrt(512)

typedef __attribute__((ext_vector_type(16))) _Float16 v16h;
typedef __attribute__((ext_vector_type(8)))  float    v8f;

// ----------------------------- entmax-1.5 ----------------------------------
// One wave (32 lanes), lane n holds slot n's raw score. Exact sort-based
// algorithm of the reference: bitonic sort + shuffle scans + ballot, wave32.
__device__ __forceinline__ float entmax15_wave(float score) {
  const int lane = threadIdx.x & 31;
  float z = score * 0.5f;
  float m = z;
  #pragma unroll
  for (int off = 16; off > 0; off >>= 1) m = fmaxf(m, __shfl_xor(m, off));
  z -= m;
  float zs = z;
  #pragma unroll
  for (int k = 2; k <= 32; k <<= 1) {
    #pragma unroll
    for (int j = k >> 1; j > 0; j >>= 1) {
      float o = __shfl_xor(zs, j);
      bool lower = (lane & j) == 0;
      bool desc  = (lane & k) == 0;
      bool take  = (lower == desc) ? (zs < o) : (zs > o);
      zs = take ? o : zs;
    }
  }
  float cs = zs, cs2 = zs * zs;
  #pragma unroll
  for (int off = 1; off < 32; off <<= 1) {
    float a = __shfl_up(cs, off);
    float b = __shfl_up(cs2, off);
    if (lane >= off) { cs += a; cs2 += b; }
  }
  float kf    = (float)(lane + 1);
  float mean  = cs / kf;
  float msq   = cs2 / kf;
  float ss    = kf * (msq - mean * mean);
  float delta = fmaxf((1.0f - ss) / kf, 0.0f);
  float tau   = mean - sqrtf(delta);
  unsigned long long ball = __ballot(tau <= zs);
  int kstar = __popcll(ball) - 1;
  float tau_star = __shfl(tau, kstar);
  float p = fmaxf(z - tau_star, 0.0f);
  return p * p;
}

// ------------------------- device-scope grid barrier ------------------------
__device__ __forceinline__ void grid_barrier(unsigned* cnt, unsigned* gen) {
  __syncthreads();                       // waits all WG memory ops, aligns waves
  if (threadIdx.x == 0) {
    __threadfence();                     // release my WG's stores device-wide
    unsigned g = __hip_atomic_load(gen, __ATOMIC_ACQUIRE, __HIP_MEMORY_SCOPE_AGENT);
    unsigned a = __hip_atomic_fetch_add(cnt, 1u, __ATOMIC_ACQ_REL, __HIP_MEMORY_SCOPE_AGENT);
    if (a == NWG - 1u) {
      __hip_atomic_store(cnt, 0u, __ATOMIC_RELAXED, __HIP_MEMORY_SCOPE_AGENT);
      __hip_atomic_fetch_add(gen, 1u, __ATOMIC_RELEASE, __HIP_MEMORY_SCOPE_AGENT);
    } else {
      while (__hip_atomic_load(gen, __ATOMIC_ACQUIRE, __HIP_MEMORY_SCOPE_AGENT) == g)
        __builtin_amdgcn_s_sleep(1);
    }
  }
  __syncthreads();
  __threadfence();                       // acquire: invalidate stale near-cache lines
}

// --------------------------- init / convert --------------------------------
__global__ __launch_bounds__(256) void init_kernel(
    const float* __restrict__ h_tape, const float* __restrict__ h_work,
    const float* __restrict__ Wh, const float* __restrict__ Wx,
    const float* __restrict__ Ww,
    _Float16* __restrict__ wh16, _Float16* __restrict__ wx16,
    _Float16* __restrict__ ww16,
    float* __restrict__ tape, float* __restrict__ work0,
    unsigned* __restrict__ bar) {
  int i = blockIdx.x * 256 + threadIdx.x;
  if (i < DIMD * DIMD) {
    wh16[i] = (_Float16)Wh[i];
    wx16[i] = (_Float16)Wx[i];
    ww16[i] = (_Float16)Ww[i];
  }
  if (i < BATCH * NSLOT * DIMD) tape[i] = h_tape[i];
  if (i < BATCH * DIMD)         work0[i] = h_work[i];
  if (i < 2)                    bar[i] = 0u;          // cnt, gen
}

// --------------------- hoisted xw = x_seq @ W_x^T + b_h --------------------
// M = B*T = 65536, N = K = 512. One wave computes a 16x64 strip (4 WMMA accums).
__global__ __launch_bounds__(256) void xw_gemm_kernel(
    const float* __restrict__ x, const _Float16* __restrict__ wx16,
    const float* __restrict__ bh, float* __restrict__ out) {
  const int tid  = threadIdx.x;
  const int lane = tid & 31;
  const int wid  = tid >> 5;
  const int s    = blockIdx.x * 8 + wid;
  const int mbase = (s >> 3) << 4;
  const int nbase = (s & 7) << 6;
  const int row = lane & 15;
  const int hh  = lane >> 4;

  v8f acc[4] = {};
  const float* arow = x + (size_t)(mbase + row) * DIMD;

  for (int kb = 0; kb < DIMD; kb += 32) {
    v16h a;
    #pragma unroll
    for (int e = 0; e < 8; ++e) {
      a[e]     = (_Float16)arow[kb + hh * 16 + e];
      a[e + 8] = (_Float16)arow[kb + hh * 16 + 8 + e];
    }
    #pragma unroll
    for (int j = 0; j < 4; ++j) {
      const _Float16* bp = wx16 + (size_t)(nbase + j * 16 + row) * DIMD + kb + hh * 16;
      v16h b;
      #pragma unroll
      for (int e = 0; e < 16; ++e) b[e] = bp[e];
      acc[j] = __builtin_amdgcn_wmma_f32_16x16x32_f16(
          false, a, false, b, (short)0, acc[j], false, false);
    }
  }
  #pragma unroll
  for (int j = 0; j < 4; ++j) {
    int n = nbase + j * 16 + row;
    float bias = bh[n];
    #pragma unroll
    for (int r = 0; r < 8; ++r) {
      int m = mbase + r + hh * 8;
      out[(size_t)m * DIMD + n] = acc[j][r] + bias;
    }
  }
}

// ---------------- per-step 32x16 WMMA tile (waves 0 and 1 only) ------------
__device__ __forceinline__ void gemm_tile(
    const float* __restrict__ A, const _Float16* __restrict__ Bw, int nb,
    float* __restrict__ hseq, int t, const float* __restrict__ readv,
    float* __restrict__ outW, int mode) {
  const int tid  = threadIdx.x;
  const int lane = tid & 31;
  const int mw   = tid >> 5;               // wave 0 -> rows 0-15, wave 1 -> 16-31
  const int row  = lane & 15;
  const int hh   = lane >> 4;

  const float*    arow = A  + (size_t)(mw * 16 + row) * DIMD;
  const _Float16* brow = Bw + (size_t)(nb + row) * DIMD;

  v8f acc = {};
  for (int kb = 0; kb < DIMD; kb += 32) {
    v16h a, b;
    #pragma unroll
    for (int e = 0; e < 8; ++e) {
      a[e]     = (_Float16)arow[kb + hh * 16 + e];
      a[e + 8] = (_Float16)arow[kb + hh * 16 + 8 + e];
    }
    const _Float16* bp = brow + kb + hh * 16;
    __builtin_prefetch(bp + 32, 0, 1);
    #pragma unroll
    for (int e = 0; e < 16; ++e) b[e] = bp[e];
    acc = __builtin_amdgcn_wmma_f32_16x16x32_f16(
        false, a, false, b, (short)0, acc, false, false);
  }
  const int n = nb + row;
  #pragma unroll
  for (int r = 0; r < 8; ++r) {
    int m = mw * 16 + r + hh * 8;          // batch index
    float v = acc[r];
    if (mode == 0) {
      size_t off = ((size_t)m * TLEN + t) * DIMD + n;
      v = tanhf(v + hseq[off] + readv[(size_t)m * DIMD + n]);
      hseq[off] = v;                       // xw consumed in place -> h_seq
    }
    outW[(size_t)m * DIMD + n] = v;
  }
}

// ------------------------ persistent scan kernel ---------------------------
__global__ __launch_bounds__(256) void scan_kernel(
    float* __restrict__ tape, float* work0, float* work1,
    float* __restrict__ readv, float* __restrict__ wvec,
    const _Float16* __restrict__ wh16, const _Float16* __restrict__ ww16,
    float* __restrict__ hseq, unsigned* __restrict__ bar) {
  __shared__ float s_wv[DIMD];
  __shared__ float s_wn[DIMD];
  __shared__ float s_sc[NSLOT];
  __shared__ float s_wt[NSLOT];

  const int b = blockIdx.x, tid = threadIdx.x, lane = tid & 31, wid = tid >> 5;
  unsigned* cnt = bar;
  unsigned* gen = bar + 1;
  float* tb = tape + (size_t)b * NSLOT * DIMD;
  float* cur = work0;
  float* nxt = work1;

  // ---- initial read vector from (tape, work0): WG b handles batch b ----
  for (int d = tid; d < DIMD; d += 256) s_wn[d] = cur[(size_t)b * DIMD + d];
  __syncthreads();
  #pragma unroll
  for (int i = 0; i < 4; ++i) {
    int n = wid * 4 + i;
    const float* tr = tb + (size_t)n * DIMD;
    float p = 0.f;
    for (int d = lane; d < DIMD; d += 32) p += tr[d] * s_wn[d];
    #pragma unroll
    for (int off = 16; off; off >>= 1) p += __shfl_xor(p, off);
    if (lane == 0) s_sc[n] = p * INV_SQRT_D;
  }
  __syncthreads();
  if (tid < 32) s_wt[tid] = entmax15_wave(s_sc[tid]);
  __syncthreads();
  for (int d = tid; d < DIMD; d += 256) {
    float acc = 0.f;
    #pragma unroll
    for (int n = 0; n < NSLOT; ++n) acc += s_wt[n] * tb[(size_t)n * DIMD + d];
    readv[(size_t)b * DIMD + d] = acc;
  }
  grid_barrier(cnt, gen);

  // ----------------------------- time loop -------------------------------
  for (int t = 0; t < TLEN; ++t) {
    // phase 1: work_new = tanh(work @ W_h^T + xw[:,t] + read), WG owns 16 cols
    if (wid < 2) gemm_tile(cur, wh16, b * 16, hseq, t, readv, nxt, 0);
    grid_barrier(cnt, gen);
    // phase 2: w_vec = work_new @ W_write^T
    if (wid < 2) gemm_tile(nxt, ww16, b * 16, nullptr, 0, nullptr, wvec, 1);
    grid_barrier(cnt, gen);

    // phase 3: WG b -> write entmax + tape update + next read (fused)
    for (int d = tid; d < DIMD; d += 256) {
      s_wv[d] = wvec[(size_t)b * DIMD + d];
      s_wn[d] = nxt[(size_t)b * DIMD + d];
    }
    __syncthreads();
    #pragma unroll
    for (int i = 0; i < 4; ++i) {
      int n = wid * 4 + i;
      const float* tr = tb + (size_t)n * DIMD;
      float p = 0.f;
      for (int d = lane; d < DIMD; d += 32) p += tr[d] * s_wv[d];
      #pragma unroll
      for (int off = 16; off; off >>= 1) p += __shfl_xor(p, off);
      if (lane == 0) s_sc[n] = p * INV_SQRT_D;
    }
    __syncthreads();
    if (tid < 32) s_wt[tid] = entmax15_wave(s_sc[tid]);
    __syncthreads();
    for (int n = 0; n < NSLOT; ++n) {
      float bb = s_wt[n];
      float* tr = tb + (size_t)n * DIMD;
      for (int d = tid; d < DIMD; d += 256)
        tr[d] = tr[d] * (1.f - bb) + bb * s_wv[d];
    }
    __syncthreads();
    #pragma unroll
    for (int i = 0; i < 4; ++i) {
      int n = wid * 4 + i;
      const float* tr = tb + (size_t)n * DIMD;
      float p = 0.f;
      for (int d = lane; d < DIMD; d += 32) p += tr[d] * s_wn[d];
      #pragma unroll
      for (int off = 16; off; off >>= 1) p += __shfl_xor(p, off);
      if (lane == 0) s_sc[n] = p * INV_SQRT_D;
    }
    __syncthreads();
    if (tid < 32) s_wt[tid] = entmax15_wave(s_sc[tid]);
    __syncthreads();
    for (int d = tid; d < DIMD; d += 256) {
      float acc = 0.f;
      #pragma unroll
      for (int n = 0; n < NSLOT; ++n) acc += s_wt[n] * tb[(size_t)n * DIMD + d];
      readv[(size_t)b * DIMD + d] = acc;
    }
    grid_barrier(cnt, gen);

    float* tmp = cur; cur = nxt; nxt = tmp;
  }

  // ---- finalize: WG b copies its tape slice and final work row block ----
  float* outTape = hseq + (size_t)BATCH * TLEN * DIMD;
  float* outLast = outTape + (size_t)BATCH * NSLOT * DIMD;
  for (int i = tid; i < NSLOT * DIMD; i += 256)
    outTape[(size_t)b * NSLOT * DIMD + i] = tb[i];
  for (int d = tid; d < DIMD; d += 256)
    outLast[(size_t)b * DIMD + d] = cur[(size_t)b * DIMD + d];
}

// ------------------------------ launcher -----------------------------------
extern "C" void kernel_launch(void* const* d_in, const int* in_sizes, int n_in,
                              void* d_out, int out_size, void* d_ws, size_t ws_size,
                              hipStream_t stream) {
  (void)in_sizes; (void)n_in; (void)out_size; (void)ws_size;
  const float* x_seq  = (const float*)d_in[0];
  const float* h_tape = (const float*)d_in[1];
  const float* h_work = (const float*)d_in[2];
  const float* W_h    = (const float*)d_in[3];
  const float* W_x    = (const float*)d_in[4];
  const float* b_h    = (const float*)d_in[5];
  const float* W_wr   = (const float*)d_in[6];
  float* out = (float*)d_out;

  // workspace layout (~3.8 MB)
  char* ws = (char*)d_ws;
  _Float16* wx16 = (_Float16*)ws;                                   // 512KB
  _Float16* wh16 = wx16 + DIMD * DIMD;                              // 512KB
  _Float16* ww16 = wh16 + DIMD * DIMD;                              // 512KB
  float* tape  = (float*)(ws + 3 * DIMD * DIMD * sizeof(_Float16)); // 2MB
  float* work0 = tape  + BATCH * NSLOT * DIMD;
  float* work1 = work0 + BATCH * DIMD;
  float* readv = work1 + BATCH * DIMD;
  float* wvec  = readv + BATCH * DIMD;
  unsigned* bar = (unsigned*)(wvec + BATCH * DIMD);                 // {cnt, gen}

  init_kernel<<<2048, 256, 0, stream>>>(h_tape, h_work, W_h, W_x, W_wr,
                                        wh16, wx16, ww16, tape, work0, bar);
  // hoisted input projection written into the h_seq output region (consumed
  // in place by the scan): 65536 x 512 x 512 f16 WMMA GEMM.
  xw_gemm_kernel<<<4096, 256, 0, stream>>>(x_seq, wx16, b_h, out);
  // whole T=2048 scan in a single persistent kernel (32 co-resident WGs).
  scan_kernel<<<NWG, 256, 0, stream>>>(tape, work0, work1, readv, wvec,
                                       wh16, ww16, out, bar);
}
